// MotifSchedulingDecoder_58626303590946
// MI455X (gfx1250) — matrix operand order repaired and checked by reading, not compiled, not run on hardware
//
#include <hip/hip_runtime.h>

// ---------------- problem constants ----------------
static constexpr int N  = 50000;   // queries (divisible by 16: 3125 tiles)
static constexpr int H  = 256;
static constexpr int L  = 128;
static constexpr int S  = 32;
static constexpr int V0 = 300;
static constexpr int V1 = 2000;

// ---------------- WMMA types ----------------
typedef __attribute__((ext_vector_type(16))) _Float16 v16h;
typedef __attribute__((ext_vector_type(8)))  _Float16 v8h;
typedef __attribute__((ext_vector_type(8)))  float    v8f;

#define WMMA_F16(a, b, c) \
  __builtin_amdgcn_wmma_f32_16x16x32_f16(false, (a), false, (b), (short)0, (c), false, false)

// A fragment (16x32 f16, M x K): lane m=lane&15 holds row m.
// lanes 0-15: K = {k0..k0+7, k0+16..k0+23}; lanes 16-31: K = {k0+8..15, k0+24..31}.
__device__ __forceinline__ v16h ldA_h(const _Float16* __restrict__ base, int strideH,
                                      int k0, int m, int hi) {
  const _Float16* p = base + (size_t)m * strideH + k0 + hi * 8;
  union { v16h v; v8h h[2]; } u;
  u.h[0] = *(const v8h*)(p);
  u.h[1] = *(const v8h*)(p + 16);
  return u.v;
}

// A fragment built from f32 memory with on-the-fly f32->f16 conversion.
__device__ __forceinline__ v16h ldA_f32(const float* __restrict__ base, int strideF,
                                        int k0, int m, int hi) {
  const float* p = base + (size_t)m * strideF + k0 + hi * 8;
  union { v16h v; _Float16 e[16]; } u;
#pragma unroll
  for (int j = 0; j < 8; ++j) u.e[j]     = (_Float16)p[j];
#pragma unroll
  for (int j = 0; j < 8; ++j) u.e[8 + j] = (_Float16)p[16 + j];
  return u.v;
}

// B fragment (32x16 f16, K x N) sourced from W stored row-major [Nout x K]
// (column n of B == row n of W). lanes 0-15: K = {k0..k0+7, k0+16..k0+23}; lanes 16-31: K = {k0+8..15, k0+24..31}.
__device__ __forceinline__ v16h ldB_h(const _Float16* __restrict__ wbase, int strideH,
                                      int k0, int n, int hi) {
  const _Float16* p = wbase + (size_t)n * strideH + k0 + hi * 8;
  union { v16h v; v8h h[2]; } u;
  u.h[0] = *(const v8h*)(p);
  u.h[1] = *(const v8h*)(p + 16);
  return u.v;
}

// ---------------- weight convert + pad (f32 -> f16, zero pad rows) ----------------
__global__ void cvt_pad_kernel(const float* __restrict__ src, _Float16* __restrict__ dst,
                               int rows, int cols, int rowsPad) {
  int total = rowsPad * cols;
  for (int i = blockIdx.x * blockDim.x + threadIdx.x; i < total;
       i += gridDim.x * blockDim.x) {
    int r = i / cols, c = i - r * cols;
    dst[i] = (r < rows) ? (_Float16)src[(size_t)r * cols + c] : (_Float16)0.f;
  }
}

// ---------------- projection GEMM: out[N,128] = f16(X[N,256] @ W[128,256]^T + b) ----
// block = 256 thr (8 waves), 16 rows per block, wave w owns col tile w (16 cols).
__global__ void proj_gemm_kernel(const float* __restrict__ X,
                                 const _Float16* __restrict__ W,   // [128 x 256] f16
                                 const float* __restrict__ bias,   // [128]
                                 _Float16* __restrict__ out) {     // [N x 128] f16
  __shared__ __attribute__((aligned(16))) _Float16 xsh[16 * 264];  // stride 264 halfs
  const int tid = threadIdx.x;
  const size_t rowbase = (size_t)blockIdx.x * 16;
  for (int i = tid; i < 16 * 256; i += 256) {
    int r = i >> 8, c = i & 255;
    xsh[r * 264 + c] = (_Float16)X[(rowbase + r) * 256 + c];
  }
  __syncthreads();
  const int wv = tid >> 5, lane = tid & 31, m = lane & 15, hi = lane >> 4;
  v8f acc = {0.f, 0.f, 0.f, 0.f, 0.f, 0.f, 0.f, 0.f};
#pragma unroll
  for (int k0 = 0; k0 < 256; k0 += 32) {
    v16h a = ldA_h(xsh, 264, k0, m, hi);
    v16h b = ldB_h(W + (size_t)wv * 16 * 256, 256, k0, m, hi);
    acc = WMMA_F16(a, b, acc);
  }
  const int col = wv * 16 + m;
  const float bv = bias[col];
#pragma unroll
  for (int r = 0; r < 8; ++r) {
    size_t row = rowbase + r + 8 * hi;
    out[row * 128 + col] = (_Float16)(acc[r] + bv);
  }
}

// ---------------- attention: per-row score/softmax/context (VALU path) ------------
// block = 256 thr = 8 waves, one wave per query row n.
__global__ void attention_kernel(const float* __restrict__ src,   // [32,32,128] f32
                                 const int* __restrict__ bidx,    // [N]
                                 const _Float16* __restrict__ q,  // [N,128] f16
                                 _Float16* __restrict__ ctx) {    // [N,128] f16
  __shared__ __attribute__((aligned(16))) _Float16 qsh[8][128];
  __shared__ float wsh[8][32];
  const int tid = threadIdx.x, wv = tid >> 5, lane = tid & 31;
  const size_t n = (size_t)blockIdx.x * 8 + wv;
#pragma unroll
  for (int j = 0; j < 4; ++j) qsh[wv][lane * 4 + j] = q[n * 128 + lane * 4 + j];
  __syncthreads();
  const int b = bidx[n];
  const float* srow = src + ((size_t)b * S + lane) * 128;
  float s = 0.f;
#pragma unroll 4
  for (int k = 0; k < 128; k += 4) {
    float4 v = *(const float4*)(srow + k);
    s += v.x * (float)qsh[wv][k]     + v.y * (float)qsh[wv][k + 1] +
         v.z * (float)qsh[wv][k + 2] + v.w * (float)qsh[wv][k + 3];
  }
  float mx = s;
  for (int off = 16; off; off >>= 1) mx = fmaxf(mx, __shfl_xor(mx, off, 32));
  float e = __expf(s - mx);
  float sum = e;
  for (int off = 16; off; off >>= 1) sum += __shfl_xor(sum, off, 32);
  wsh[wv][lane] = e / sum;
  __syncthreads();
  const int c0 = lane * 4;
  float a0 = 0.f, a1 = 0.f, a2 = 0.f, a3 = 0.f;
#pragma unroll 4
  for (int si = 0; si < S; ++si) {
    float w = wsh[wv][si];
    float4 v = *(const float4*)(src + ((size_t)b * S + si) * 128 + c0);
    a0 += w * v.x; a1 += w * v.y; a2 += w * v.z; a3 += w * v.w;
  }
  _Float16* o = ctx + n * 128 + c0;
  o[0] = (_Float16)a0; o[1] = (_Float16)a1; o[2] = (_Float16)a2; o[3] = (_Float16)a3;
}

// ---------------- fused 2-layer MLP ----------------
// x = [act(f32,N x 256) | ctx(f16,N x 128)]; h = relu(x@W1^T+b1) [N,256] in LDS;
// out = h@W2^T+b2 [N,Vout] f32 written straight to d_out.
__global__ void fused_mlp_kernel(const float* __restrict__ act,
                                 const _Float16* __restrict__ ctx,
                                 const _Float16* __restrict__ W1,  // [256 x 384] f16
                                 const float* __restrict__ b1,     // [256]
                                 const _Float16* __restrict__ W2,  // [VPad x 256] f16
                                 const float* __restrict__ b2,     // [Vout]
                                 float* __restrict__ out,          // [N x Vout]
                                 int Vout, int VPad) {
  __shared__ __attribute__((aligned(16))) _Float16 xsh[16 * 392];  // 16 x 384, pad 8
  __shared__ __attribute__((aligned(16))) _Float16 hsh[16 * 264];  // 16 x 256, pad 8
  const int tid = threadIdx.x;
  const size_t rowbase = (size_t)blockIdx.x * 16;
  for (int i = tid; i < 16 * 256; i += 256) {
    int r = i >> 8, c = i & 255;
    xsh[r * 392 + c] = (_Float16)act[(rowbase + r) * 256 + c];
  }
  for (int i = tid; i < 16 * 128; i += 256) {
    int r = i >> 7, c = i & 127;
    xsh[r * 392 + 256 + c] = ctx[(rowbase + r) * 128 + c];
  }
  __syncthreads();
  const int wv = tid >> 5, lane = tid & 31, m = lane & 15, hi = lane >> 4;

  // ---- GEMM1: 16 col tiles over 8 waves (2 each), K = 384 ----
  const int ct0 = wv * 2, ct1 = ct0 + 1;
  v8f acc0 = {0.f, 0.f, 0.f, 0.f, 0.f, 0.f, 0.f, 0.f};
  v8f acc1 = {0.f, 0.f, 0.f, 0.f, 0.f, 0.f, 0.f, 0.f};
#pragma unroll
  for (int k0 = 0; k0 < 384; k0 += 32) {
    v16h a  = ldA_h(xsh, 392, k0, m, hi);
    v16h bA = ldB_h(W1 + (size_t)ct0 * 16 * 384, 384, k0, m, hi);
    acc0 = WMMA_F16(a, bA, acc0);
    v16h bB = ldB_h(W1 + (size_t)ct1 * 16 * 384, 384, k0, m, hi);
    acc1 = WMMA_F16(a, bB, acc1);
  }
  {
    int colA = ct0 * 16 + m, colB = ct1 * 16 + m;
    float bvA = b1[colA], bvB = b1[colB];
#pragma unroll
    for (int r = 0; r < 8; ++r) {
      float hA = acc0[r] + bvA; hA = hA > 0.f ? hA : 0.f;
      float hB = acc1[r] + bvB; hB = hB > 0.f ? hB : 0.f;
      hsh[(r + 8 * hi) * 264 + colA] = (_Float16)hA;
      hsh[(r + 8 * hi) * 264 + colB] = (_Float16)hB;
    }
  }
  __syncthreads();

  // ---- GEMM2: Vout cols (padded to VPad), K = 256, tiles round-robin on waves ----
  const int ntiles = VPad >> 4;
  for (int ct = wv; ct < ntiles; ct += 8) {
    if (ct + 8 < ntiles)  // prefetch next weight tile -> global_prefetch_b8
      __builtin_prefetch((const void*)(W2 + (size_t)(ct + 8) * 16 * 256), 0, 1);
    v8f acc = {0.f, 0.f, 0.f, 0.f, 0.f, 0.f, 0.f, 0.f};
#pragma unroll
    for (int k0 = 0; k0 < 256; k0 += 32) {
      v16h a = ldA_h(hsh, 264, k0, m, hi);
      v16h b = ldB_h(W2 + (size_t)ct * 16 * 256, 256, k0, m, hi);
      acc = WMMA_F16(a, b, acc);
    }
    const int col = ct * 16 + m;
    if (col < Vout) {
      const float bv = b2[col];
#pragma unroll
      for (int r = 0; r < 8; ++r) {
        size_t row = rowbase + r + 8 * hi;
        out[row * (size_t)Vout + col] = acc[r] + bv;
      }
    }
  }
}

// ---------------- assm final: out[n] = sum_l p[n,l]*c[n,l] ----------------
__global__ void mulsum_kernel(const _Float16* __restrict__ p,
                              const _Float16* __restrict__ c,
                              float* __restrict__ out) {
  const int tid = threadIdx.x, wv = tid >> 5, lane = tid & 31;
  const size_t n = (size_t)blockIdx.x * 8 + wv;
  const int k = lane * 4;
  float s = 0.f;
#pragma unroll
  for (int j = 0; j < 4; ++j)
    s += (float)p[n * 128 + k + j] * (float)c[n * 128 + k + j];
  for (int off = 16; off; off >>= 1) s += __shfl_xor(s, off, 32);
  if (lane == 0) out[n] = s;
}

// ---------------- host side ----------------
extern "C" void kernel_launch(void* const* d_in, const int* in_sizes, int n_in,
                              void* d_out, int out_size, void* d_ws, size_t ws_size,
                              hipStream_t stream) {
  (void)in_sizes; (void)n_in; (void)out_size; (void)ws_size;

  const float* src_tree  = (const float*)d_in[0];
  const float* src_graph = (const float*)d_in[1];
  const int*   bidx      = (const int*)d_in[2];
  const float* topo_vecs = (const float*)d_in[3];
  const float* cls_vecs  = (const float*)d_in[4];
  const float* assm_vecs = (const float*)d_in[5];
  const float* A_topo_w  = (const float*)d_in[6];
  const float* A_topo_b  = (const float*)d_in[7];
  const float* A_cls_w   = (const float*)d_in[8];
  const float* A_cls_b   = (const float*)d_in[9];
  const float* A_assm_w  = (const float*)d_in[10];
  const float* A_assm_b  = (const float*)d_in[11];
  const float* topo_w1   = (const float*)d_in[12];
  const float* topo_b1   = (const float*)d_in[13];
  const float* topo_w2   = (const float*)d_in[14];
  const float* topo_b2   = (const float*)d_in[15];
  const float* cls_w1    = (const float*)d_in[16];
  const float* cls_b1    = (const float*)d_in[17];
  const float* cls_w2    = (const float*)d_in[18];
  const float* cls_b2    = (const float*)d_in[19];
  const float* icls_w1   = (const float*)d_in[20];
  const float* icls_b1   = (const float*)d_in[21];
  const float* icls_w2   = (const float*)d_in[22];
  const float* icls_b2   = (const float*)d_in[23];
  const float* assm_w    = (const float*)d_in[24];
  const float* assm_b    = (const float*)d_in[25];

  float* out_topo = (float*)d_out;
  float* out_cls  = out_topo + N;
  float* out_icls = out_cls + (size_t)N * V0;
  float* out_assm = out_icls + (size_t)N * V1;

  // workspace carve (256B aligned slices)
  size_t off = 0;
  auto take = [&](size_t halfs) -> _Float16* {
    _Float16* ptr = (_Float16*)((char*)d_ws + off);
    off += (halfs * sizeof(_Float16) + 255) & ~(size_t)255;
    return ptr;
  };
  _Float16* wA_topo = take(128 * 256);
  _Float16* wA_cls  = take(128 * 256);
  _Float16* wA_assm = take(128 * 256);
  _Float16* w1_topo = take(256 * 384);
  _Float16* w1_cls  = take(256 * 384);
  _Float16* w1_icls = take(256 * 384);
  _Float16* w2_topo = take(16 * 256);       // padded 1 -> 16 rows
  _Float16* w2_cls  = take(304 * 256);      // padded 300 -> 304 rows
  _Float16* w2_icls = take((size_t)V1 * 256);
  _Float16* w_assm  = take(128 * 256);
  _Float16* qbuf    = take((size_t)N * 128);  // reused across the 3 paths
  _Float16* cbuf    = take((size_t)N * 128);  // reused ctx buffer
  _Float16* pbuf    = take((size_t)N * 128);  // assm projection

  // 1) convert/pad weights to f16
  struct CvtJob { const float* s; _Float16* d; int r, c, rp; };
  const CvtJob jobs[] = {
      {A_topo_w, wA_topo, 128, 256, 128},  {A_cls_w, wA_cls, 128, 256, 128},
      {A_assm_w, wA_assm, 128, 256, 128},  {topo_w1, w1_topo, 256, 384, 256},
      {cls_w1, w1_cls, 256, 384, 256},     {icls_w1, w1_icls, 256, 384, 256},
      {topo_w2, w2_topo, 1, 256, 16},      {cls_w2, w2_cls, 300, 256, 304},
      {icls_w2, w2_icls, V1, 256, V1},     {assm_w, w_assm, 128, 256, 128},
  };
  for (const CvtJob& j : jobs) {
    int total = j.rp * j.c;
    int blocks = (total + 255) / 256;
    if (blocks > 1024) blocks = 1024;
    cvt_pad_kernel<<<blocks, 256, 0, stream>>>(j.s, j.d, j.r, j.c, j.rp);
  }

  const dim3 blk(256);
  const int gTile = N / 16;  // 3125
  const int gRow  = N / 8;   // 6250

  // 2) topo path
  proj_gemm_kernel<<<gTile, blk, 0, stream>>>(topo_vecs, wA_topo, A_topo_b, qbuf);
  attention_kernel<<<gRow, blk, 0, stream>>>(src_tree, bidx, qbuf, cbuf);
  fused_mlp_kernel<<<gTile, blk, 0, stream>>>(topo_vecs, cbuf, w1_topo, topo_b1,
                                              w2_topo, topo_b2, out_topo, 1, 16);
  // 3) cls / icls path (shared attention context)
  proj_gemm_kernel<<<gTile, blk, 0, stream>>>(cls_vecs, wA_cls, A_cls_b, qbuf);
  attention_kernel<<<gRow, blk, 0, stream>>>(src_tree, bidx, qbuf, cbuf);
  fused_mlp_kernel<<<gTile, blk, 0, stream>>>(cls_vecs, cbuf, w1_cls, cls_b1,
                                              w2_cls, cls_b2, out_cls, V0, 304);
  fused_mlp_kernel<<<gTile, blk, 0, stream>>>(cls_vecs, cbuf, w1_icls, icls_b1,
                                              w2_icls, icls_b2, out_icls, V1, V1);
  // 4) assm path
  proj_gemm_kernel<<<gTile, blk, 0, stream>>>(assm_vecs, wA_assm, A_assm_b, qbuf);
  attention_kernel<<<gRow, blk, 0, stream>>>(src_graph, bidx, qbuf, cbuf);
  proj_gemm_kernel<<<gTile, blk, 0, stream>>>(assm_vecs, w_assm, assm_b, pbuf);
  mulsum_kernel<<<gRow, blk, 0, stream>>>(pbuf, cbuf, out_assm);
}
